// Attention_15710990368873
// MI455X (gfx1250) — matrix operand order repaired and checked by reading, not compiled, not run on hardware
//
#include <hip/hip_runtime.h>
#include <hip/hip_bf16.h>

typedef __attribute__((ext_vector_type(16))) __bf16 bf16x16;
typedef __attribute__((ext_vector_type(8)))  __bf16 bf16x8;
typedef __attribute__((ext_vector_type(8)))  float  f32x8;
typedef __attribute__((ext_vector_type(4)))  float  f32x4;

__device__ __forceinline__ bf16x16 join8(bf16x8 a, bf16x8 b) {
    bf16x16 r;
#pragma unroll
    for (int i = 0; i < 8; ++i) { r[i] = a[i]; r[i + 8] = b[i]; }
    return r;
}

// A fragment (16x32 bf16, lane = row M, ISA 7.12.2): row-major source, ld = ldk.
__device__ __forceinline__ bf16x16 load_frag_a(const __bf16* base, int ldk, int lane) {
    int m = lane & 15, half = lane >> 4;
    const __bf16* p = base + m * ldk + half * 8;
    bf16x8 lo = *(const bf16x8*)p;
    bf16x8 hi = *(const bf16x8*)(p + 16);
    return join8(lo, hi);
}

// B fragment (32x16 bf16, lane = col N): source laid out [N][K] (K contiguous).
__device__ __forceinline__ bf16x16 load_frag_b(const __bf16* base, int ldk, int lane) {
    int n = lane & 15, half = lane >> 4;
    const __bf16* p = base + n * ldk + half * 8;
    bf16x8 lo = *(const bf16x8*)p;
    bf16x8 hi = *(const bf16x8*)(p + 16);
    return join8(lo, hi);
}

__device__ __forceinline__ f32x8 wmma_bf16(bf16x16 a, bf16x16 b, f32x8 c) {
    return __builtin_amdgcn_wmma_f32_16x16x32_bf16(
        false, a, false, b, (short)0, c, false, false);
}

#define BM 128
#define BN 128
#define BK 32
#define KP 40  // padded K ld (80 bytes/row, multiple of 16B)

// C = A[M][K] @ B[K][N] + bias. qkv_mode=1: scatter bf16 into per-head Q/K/V buffers.
// Software-pipelined: tile k+1 global loads are in flight during tile k WMMAs.
__global__ __launch_bounds__(256) void gemm_bf16_kernel(
    const float* __restrict__ A, const float* __restrict__ Bw,
    const float* __restrict__ bias, int M, int N, int K,
    int qkv_mode, float* __restrict__ Cout,
    __bf16* __restrict__ Qb, __bf16* __restrict__ Kb, __bf16* __restrict__ Vb) {
    __shared__ __align__(16) __bf16 As[BM * KP];
    __shared__ __align__(16) __bf16 Bs[BN * KP];

    int tid = threadIdx.x;
    int lane = tid & 31, wid = tid >> 5;
    int waveM = wid >> 1, waveN = wid & 1;   // 4x2 waves, 32x64 per wave
    int bm = blockIdx.y * BM, bn = blockIdx.x * BN;

    f32x4 ra[4], rb[4];  // register-staged A (128x32) and B (32x128) tiles

#define LOAD_TILES(k0_)                                                            \
    _Pragma("unroll")                                                              \
    for (int it = 0; it < 4; ++it) {                                               \
        int idx = tid + it * 256;                                                  \
        ra[it] = *(const f32x4*)(A + (size_t)(bm + (idx >> 3)) * K + (k0_) +       \
                                 ((idx & 7) << 2));                                \
        rb[it] = *(const f32x4*)(Bw + (size_t)((k0_) + (idx >> 5)) * N + bn +      \
                                 ((idx & 31) << 2));                               \
    }

    LOAD_TILES(0)
    f32x8 acc[2][4] = {};

    for (int k0 = 0; k0 < K; k0 += BK) {
        __syncthreads();  // previous tile's readers are done
#pragma unroll
        for (int it = 0; it < 4; ++it) {
            int idx = tid + it * 256;
            int r = idx >> 3, c4 = (idx & 7) << 2;
            __bf16* d = As + r * KP + c4;
            d[0] = (__bf16)ra[it][0]; d[1] = (__bf16)ra[it][1];
            d[2] = (__bf16)ra[it][2]; d[3] = (__bf16)ra[it][3];
            int kk = idx >> 5, n4 = (idx & 31) << 2;
            Bs[(n4 + 0) * KP + kk] = (__bf16)rb[it][0];
            Bs[(n4 + 1) * KP + kk] = (__bf16)rb[it][1];
            Bs[(n4 + 2) * KP + kk] = (__bf16)rb[it][2];
            Bs[(n4 + 3) * KP + kk] = (__bf16)rb[it][3];
        }
        __syncthreads();
        if (k0 + BK < K) { LOAD_TILES(k0 + BK) }  // overlap with WMMA below

        bf16x16 af[2], bfr[4];
#pragma unroll
        for (int tm = 0; tm < 2; ++tm)
            af[tm] = load_frag_a(As + (waveM * 32 + tm * 16) * KP, KP, lane);
#pragma unroll
        for (int tn = 0; tn < 4; ++tn)
            bfr[tn] = load_frag_b(Bs + (waveN * 64 + tn * 16) * KP, KP, lane);
#pragma unroll
        for (int tm = 0; tm < 2; ++tm)
#pragma unroll
            for (int tn = 0; tn < 4; ++tn)
                acc[tm][tn] = wmma_bf16(af[tm], bfr[tn], acc[tm][tn]);
    }
#undef LOAD_TILES

    int half = lane >> 4, n = lane & 15;
#pragma unroll
    for (int tm = 0; tm < 2; ++tm)
#pragma unroll
        for (int tn = 0; tn < 4; ++tn)
#pragma unroll
            for (int r = 0; r < 8; ++r) {
                int row = bm + waveM * 32 + tm * 16 + half * 8 + r;
                int col = bn + waveN * 64 + tn * 16 + n;
                float v = acc[tm][tn][r] + bias[col];
                if (qkv_mode) {
                    int t = col >> 10, d = col & 1023;
                    int h = d >> 6, dh = d & 63;
                    int b = row >> 12, l = row & 4095;
                    size_t dst = (((size_t)(b * 16 + h)) * 4096 + l) * 64 + dh;
                    __bf16 bv = (__bf16)v;
                    if (t == 0) Qb[dst] = bv;
                    else if (t == 1) Kb[dst] = bv;
                    else Vb[dst] = bv;
                } else {
                    Cout[(size_t)row * N + col] = v;
                }
            }
}

// In-place RoPE on bf16 Q and K in [B*H][L][64] layout. One thread per (bh,l,pair).
__global__ __launch_bounds__(256) void rope_kernel(__bf16* __restrict__ Q,
                                                   __bf16* __restrict__ K, int total) {
    int idx = blockIdx.x * blockDim.x + threadIdx.x;
    if (idx >= total) return;
    int dh = idx & 31;
    int l  = (idx >> 5) & 4095;
    int bh = idx >> 17;
    size_t base = ((size_t)bh * 4096 + l) * 64;
    float angle = (float)l * __powf(500.0f, -(float)dh * (1.0f / 32.0f));
    float s, c;
    __sincosf(angle, &s, &c);
    float q1 = (float)Q[base + dh], q2 = (float)Q[base + dh + 32];
    Q[base + dh]      = (__bf16)(q1 * c - q2 * s);
    Q[base + dh + 32] = (__bf16)(q1 * s + q2 * c);
    float k1 = (float)K[base + dh], k2 = (float)K[base + dh + 32];
    K[base + dh]      = (__bf16)(k1 * c - k2 * s);
    K[base + dh + 32] = (__bf16)(k1 * s + k2 * c);
}

#define FQ 128
#define FKT 64
#define FP 72  // padded ld (144 bytes, multiple of 16B)

// Flash attention per (b,h): Q[L,64] bf16, K[L,64] bf16, V[L,64] bf16 -> O fp32 [B][L][H*64]
__global__ __launch_bounds__(256) void flash_kernel(
    const __bf16* __restrict__ Qb, const __bf16* __restrict__ Kg_,
    const __bf16* __restrict__ Vg_, float* __restrict__ O) {
    __shared__ __align__(16) __bf16 Ks[FKT * FP];       // [key][dh]
    __shared__ __align__(16) __bf16 Vt[FKT * FP];       // [dh][key]
    __shared__ __align__(16) __bf16 Pl[8 * 16 * FP];    // per-wave P (16 x 64)

    int tid = threadIdx.x, lane = tid & 31, wid = tid >> 5;
    int half = lane >> 4, n = lane & 15;
    int bh = blockIdx.y;
    int q0 = blockIdx.x * FQ + wid * 16;

    const __bf16* Qp = Qb  + ((size_t)bh * 4096 + q0) * 64;
    const __bf16* Kg = Kg_ + (size_t)bh * 4096 * 64;
    const __bf16* Vg = Vg_ + (size_t)bh * 4096 * 64;

    // Q fragments held in registers across all key tiles (dh chunks 0..31, 32..63)
    bf16x16 qf[2];
#pragma unroll
    for (int kk = 0; kk < 2; ++kk) qf[kk] = load_frag_a(Qp + kk * 32, 64, lane);

    f32x8 o[4] = {};
    float mrow[8], lrow[8];
#pragma unroll
    for (int r = 0; r < 8; ++r) { mrow[r] = -1e30f; lrow[r] = 0.0f; }
    __bf16* Pw = Pl + wid * 16 * FP;

    for (int kt = 0; kt < 4096; kt += FKT) {
        // Batch V-tile loads into registers (in flight across the barrier)
        unsigned vv[8];
#pragma unroll
        for (int it = 0; it < 8; ++it) {
            int idx = tid + it * 256;
            int row = idx >> 5, c2 = (idx & 31) << 1;
            vv[it] = *(const unsigned*)(Vg + (size_t)(kt + row) * 64 + c2);
        }
        __syncthreads();  // all waves done reading previous Ks/Vt

        // K tile: bf16 -> LDS unchanged, via CDNA5 async copy (ASYNCcnt, no VGPR data)
#pragma unroll
        for (int it = 0; it < 2; ++it) {
            int idx = tid + it * 256;
            int row = idx >> 3, c8 = (idx & 7) << 3;
            const __bf16* g = Kg + (size_t)(kt + row) * 64 + c8;
            unsigned ldso = (unsigned)(size_t)(Ks + row * FP + c8);  // low 32b = LDS offset
            asm volatile("global_load_async_to_lds_b128 %0, %1, off"
                         :: "v"(ldso), "v"(g) : "memory");
        }
        // V transpose scatter from registers
#pragma unroll
        for (int it = 0; it < 8; ++it) {
            int idx = tid + it * 256;
            int row = idx >> 5, c2 = (idx & 31) << 1;
            __bf16 v0 = ((const __bf16*)&vv[it])[0], v1 = ((const __bf16*)&vv[it])[1];
            Vt[(c2 + 0) * FP + row] = v0;
            Vt[(c2 + 1) * FP + row] = v1;
        }
        asm volatile("s_wait_asynccnt 0" ::: "memory");
        __syncthreads();

        // S = Q @ K^T  (16 queries x 64 keys per wave)
        f32x8 s[4] = {};
#pragma unroll
        for (int tn = 0; tn < 4; ++tn)
#pragma unroll
            for (int kk = 0; kk < 2; ++kk) {
                bf16x16 bfrag = load_frag_b(Ks + (tn * 16) * FP + kk * 32, FP, lane);
                s[tn] = wmma_bf16(qf[kk], bfrag, s[tn]);
            }

        // Online softmax: each row lives in a 16-lane group at vgpr slot r
#pragma unroll
        for (int r = 0; r < 8; ++r) {
            float v0 = s[0][r] * 0.125f, v1 = s[1][r] * 0.125f;
            float v2 = s[2][r] * 0.125f, v3 = s[3][r] * 0.125f;
            float mx = fmaxf(fmaxf(v0, v1), fmaxf(v2, v3));
            for (int off = 1; off < 16; off <<= 1)
                mx = fmaxf(mx, __shfl_xor(mx, off, 32));
            float mnew = fmaxf(mrow[r], mx);
            float alpha = __expf(mrow[r] - mnew);
            float p0 = __expf(v0 - mnew), p1 = __expf(v1 - mnew);
            float p2 = __expf(v2 - mnew), p3 = __expf(v3 - mnew);
            float ps = p0 + p1 + p2 + p3;
            for (int off = 1; off < 16; off <<= 1)
                ps += __shfl_xor(ps, off, 32);
            lrow[r] = lrow[r] * alpha + ps;
            mrow[r] = mnew;
            o[0][r] *= alpha; o[1][r] *= alpha; o[2][r] *= alpha; o[3][r] *= alpha;
            int prow = r + half * 8;
            Pw[prow * FP +  0 + n] = (__bf16)p0;
            Pw[prow * FP + 16 + n] = (__bf16)p1;
            Pw[prow * FP + 32 + n] = (__bf16)p2;
            Pw[prow * FP + 48 + n] = (__bf16)p3;
        }
        asm volatile("s_wait_dscnt 0" ::: "memory");  // P stores visible before A-frag gather

        // O += P @ V  (P: 16x64 keys as A-frags, V^T staged [dh][key] as B-frags)
#pragma unroll
        for (int kk = 0; kk < 2; ++kk) {
            bf16x16 pf = load_frag_a(Pw + kk * 32, FP, lane);
#pragma unroll
            for (int tn = 0; tn < 4; ++tn) {
                bf16x16 vf = load_frag_b(Vt + (tn * 16) * FP + kk * 32, FP, lane);
                o[tn] = wmma_bf16(pf, vf, o[tn]);
            }
        }
    }

    int b = bh >> 4, h = bh & 15;
#pragma unroll
    for (int tn = 0; tn < 4; ++tn)
#pragma unroll
        for (int r = 0; r < 8; ++r) {
            int qrow = q0 + half * 8 + r;
            int col = h * 64 + tn * 16 + n;
            O[((size_t)b * 4096 + qrow) * 1024 + col] = o[tn][r] / lrow[r];
        }
}

extern "C" void kernel_launch(void* const* d_in, const int* in_sizes, int n_in,
                              void* d_out, int out_size, void* d_ws, size_t ws_size,
                              hipStream_t stream) {
    (void)in_sizes; (void)n_in; (void)out_size; (void)ws_size;
    const float* x     = (const float*)d_in[0];
    const float* w_qkv = (const float*)d_in[1];
    const float* b_qkv = (const float*)d_in[2];
    const float* w_out = (const float*)d_in[3];
    const float* b_out = (const float*)d_in[4];
    float* out = (float*)d_out;

    char* ws = (char*)d_ws;
    const size_t QKV_ELEMS = (size_t)2 * 16 * 4096 * 64;  // 8,388,608
    __bf16* Qb = (__bf16*)(ws);
    __bf16* Kb = (__bf16*)(ws + QKV_ELEMS * 2);
    __bf16* Vb = (__bf16*)(ws + QKV_ELEMS * 4);
    float* attO = (float*)(ws + QKV_ELEMS * 6);           // 32 MB fp32

    // 1) QKV projection: 8192 x 3072 x 1024, scatter bf16 per-head
    gemm_bf16_kernel<<<dim3(3072 / BN, 8192 / BM), 256, 0, stream>>>(
        x, w_qkv, b_qkv, 8192, 3072, 1024, 1, nullptr, Qb, Kb, Vb);

    // 2) RoPE in-place on Q and K (B*H*L*32 pairs)
    const int pairs = 2 * 16 * 4096 * 32;
    rope_kernel<<<(pairs + 255) / 256, 256, 0, stream>>>(Qb, Kb, pairs);

    // 3) Flash attention: 32 q-tiles x 32 (b,h)
    flash_kernel<<<dim3(32, 32), 256, 0, stream>>>(Qb, Kb, Vb, attO);

    // 4) Output projection: 8192 x 1024 x 1024 -> d_out fp32
    gemm_bf16_kernel<<<dim3(1024 / BN, 8192 / BM), 256, 0, stream>>>(
        attO, w_out, b_out, 8192, 1024, 1024, 0, out, nullptr, nullptr, nullptr);
}
